// RateDistortionLoss_73950746902747
// MI455X (gfx1250) — matrix operand order repaired and checked by reading, not compiled, not run on hardware
//
#include <hip/hip_runtime.h>
#include <math.h>

typedef __attribute__((ext_vector_type(8)))  float  v8f;
typedef __attribute__((ext_vector_type(16))) __bf16 v16bf;
typedef __attribute__((ext_vector_type(8)))  __bf16 v8bf;
typedef __attribute__((ext_vector_type(4)))  __bf16 v4bf;
typedef __attribute__((ext_vector_type(4)))  int    v4i;

#define BM 128
#define BN 128
#define BK 32

// --------------------------- CDNA5 async-to-LDS path -----------------------
#if defined(__AMDGCN__)
#  if __has_builtin(__builtin_amdgcn_global_load_async_to_lds_b128)
#    define ASYNC_STAGE 1
#  else
#    warning "gfx1250 global_load_async_to_lds builtin not found; using sync LDS staging"
#    define ASYNC_STAGE 0
#  endif
#else
#  define ASYNC_STAGE 0
#endif

// builtin signature (probe-confirmed): arg0 is v4i* in addrspace(1)
typedef __attribute__((address_space(1))) v4i gas_v4i;
typedef __attribute__((address_space(3))) v4i las_v4i;

__device__ __forceinline__ void async_copy_b128(const __bf16* g, __bf16* l) {
#if ASYNC_STAGE
    __builtin_amdgcn_global_load_async_to_lds_b128(
        (gas_v4i*)g, (las_v4i*)l, /*offset=*/0, /*cpol=*/0);
#else
    (void)g; (void)l;
#endif
}

template <int N>
__device__ __forceinline__ void wait_asynccnt() {
#if defined(__AMDGCN__)
#  if __has_builtin(__builtin_amdgcn_s_wait_asynccnt)
    __builtin_amdgcn_s_wait_asynccnt(N);
#  else
    asm volatile("s_wait_asynccnt %0" ::"i"(N));
#  endif
#endif
}

// ---------------------------------------------------------------------------
// Pre-pass: trace(H) partials (one partial per block, deterministic)
// ---------------------------------------------------------------------------
__global__ __launch_bounds__(256) void rdl_trace_kernel(const float* __restrict__ H,
                                                        float* __restrict__ partial,
                                                        int D) {
    int i = blockIdx.x * 256 + threadIdx.x;
    float v = (i < D) ? H[(size_t)i * D + i] : 0.0f;
    #pragma unroll
    for (int off = 16; off > 0; off >>= 1) v += __shfl_xor(v, off, 32);
    __shared__ float ws[8];
    int lane = threadIdx.x & 31, wid = threadIdx.x >> 5;
    if (lane == 0) ws[wid] = v;
    __syncthreads();
    if (threadIdx.x == 0) {
        float s = 0.0f;
        #pragma unroll
        for (int w = 0; w < 8; ++w) s += ws[w];
        partial[blockIdx.x] = s;
    }
}

// ---------------------------------------------------------------------------
// Pre-pass: sum(log(likelihoods)) partials, 4 elems/thread, float4 loads
// ---------------------------------------------------------------------------
__global__ __launch_bounds__(256) void rdl_logsum_kernel(const float* __restrict__ lik,
                                                         float* __restrict__ partial) {
    size_t i = ((size_t)blockIdx.x * 256 + threadIdx.x) * 4;
    float4 x = *(const float4*)&lik[i];
    float v = __logf(x.x) + __logf(x.y) + __logf(x.z) + __logf(x.w);
    #pragma unroll
    for (int off = 16; off > 0; off >>= 1) v += __shfl_xor(v, off, 32);
    __shared__ float ws[8];
    int lane = threadIdx.x & 31, wid = threadIdx.x >> 5;
    if (lane == 0) ws[wid] = v;
    __syncthreads();
    if (threadIdx.x == 0) {
        float s = 0.0f;
        #pragma unroll
        for (int w = 0; w < 8; ++w) s += ws[w];
        partial[blockIdx.x] = s;
    }
}

// ---------------------------------------------------------------------------
// Pre-pass: diff = x_hat - ori_W, converted to bf16 (4 elems/thread)
// ---------------------------------------------------------------------------
__global__ __launch_bounds__(256) void rdl_diff_bf16_kernel(const float* __restrict__ xhat,
                                                            const float* __restrict__ w,
                                                            __bf16* __restrict__ dA) {
    size_t i = ((size_t)blockIdx.x * 256 + threadIdx.x) * 4;
    float4 a = *(const float4*)&xhat[i];
    float4 b = *(const float4*)&w[i];
    v4bf o;
    o[0] = (__bf16)(a.x - b.x);
    o[1] = (__bf16)(a.y - b.y);
    o[2] = (__bf16)(a.z - b.z);
    o[3] = (__bf16)(a.w - b.w);
    *(v4bf*)&dA[i] = o;
}

// ---------------------------------------------------------------------------
// Pre-pass: Bt[n*D + k] = (bf16)H[k*D + n]  (LDS tile transpose, 32x32 tiles)
// ---------------------------------------------------------------------------
__global__ __launch_bounds__(256) void rdl_transpose_bf16_kernel(const float* __restrict__ H,
                                                                 __bf16* __restrict__ Bt,
                                                                 int D) {
    __shared__ float tile[32][33];
    int x  = blockIdx.x * 32 + threadIdx.x;
    int y0 = blockIdx.y * 32;
    #pragma unroll
    for (int i = threadIdx.y; i < 32; i += 8)
        tile[i][threadIdx.x] = H[(size_t)(y0 + i) * D + x];
    __syncthreads();
    int xt  = blockIdx.y * 32 + threadIdx.x;
    int y0t = blockIdx.x * 32;
    #pragma unroll
    for (int i = threadIdx.y; i < 32; i += 8)
        Bt[(size_t)(y0t + i) * D + xt] = (__bf16)tile[threadIdx.x][i];
}

// ---------------------------------------------------------------------------
// Fused GEMM: per block, a 128x128 tile of T = diff @ H via bf16 WMMA, then
// the epilogue accumulates sum(T_tile * diff_tile) into one partial scalar.
// Double-buffered LDS staging via GLOBAL_LOAD_ASYNC_TO_LDS_B128 (ASYNCcnt),
// branchless steady-state loop (last K-step peeled).
// 8 waves in a 4x2 grid; each wave owns 32x64 = 2x4 WMMA tiles.
// ---------------------------------------------------------------------------
__global__ __launch_bounds__(256) void rdl_gemm_quad_kernel(const __bf16* __restrict__ A,   // diff [N,D] bf16
                                                            const __bf16* __restrict__ Bt,  // H^T  [D,D] bf16
                                                            float* __restrict__ partial,
                                                            int D) {
    __shared__ __align__(16) __bf16 sA[2][BM * BK];  // [BM][BK] K-contiguous
    __shared__ __align__(16) __bf16 sB[2][BN * BK];  // [BN][BK] K-contiguous (B = H^T)

    const int tid  = threadIdx.x;
    const int lane = tid & 31;
    const int wid  = tid >> 5;
    const int half = lane >> 4;
    const int l15  = lane & 15;

    const int block_m = blockIdx.x * BM;
    const int block_n = blockIdx.y * BN;
    const int wave_m  = (wid & 3) << 5;   // 0,32,64,96
    const int wave_n  = (wid >> 2) << 6;  // 0,64

    // per-thread staging chunks (b128 = 8 bf16). A: 512 chunks, B: 512 chunks.
    const int r0 = tid >> 2;             // rows 0..63   (chunk 0)
    const int r1 = r0 + 64;              // rows 64..127 (chunk 1)
    const int cc = (tid & 3) * 8;        // K column within tile

    auto stage = [&](int buf, int k0) {
#if ASYNC_STAGE
        async_copy_b128(&A [(size_t)(block_m + r0) * D + k0 + cc], &sA[buf][r0 * BK + cc]);
        async_copy_b128(&A [(size_t)(block_m + r1) * D + k0 + cc], &sA[buf][r1 * BK + cc]);
        async_copy_b128(&Bt[(size_t)(block_n + r0) * D + k0 + cc], &sB[buf][r0 * BK + cc]);
        async_copy_b128(&Bt[(size_t)(block_n + r1) * D + k0 + cc], &sB[buf][r1 * BK + cc]);
#else
        v8bf a0 = *(const v8bf*)&A [(size_t)(block_m + r0) * D + k0 + cc];
        v8bf a1 = *(const v8bf*)&A [(size_t)(block_m + r1) * D + k0 + cc];
        v8bf b0 = *(const v8bf*)&Bt[(size_t)(block_n + r0) * D + k0 + cc];
        v8bf b1 = *(const v8bf*)&Bt[(size_t)(block_n + r1) * D + k0 + cc];
        *(v8bf*)&sA[buf][r0 * BK + cc] = a0;
        *(v8bf*)&sA[buf][r1 * BK + cc] = a1;
        *(v8bf*)&sB[buf][r0 * BK + cc] = b0;
        *(v8bf*)&sB[buf][r1 * BK + cc] = b1;
#endif
    };

    v8f acc[2][4] = {};

    auto compute = [&](int buf) {
        // A 16x32 frag: lane half g holds K = g*8+{0..7} and 16+g*8+{0..7}, M = l15
        v16bf afrag[2];
        #pragma unroll
        for (int i = 0; i < 2; ++i) {
            int m = wave_m + i * 16 + l15;
            v8bf lo = *(const v8bf*)&sA[buf][m * BK + half * 8];
            v8bf hi = *(const v8bf*)&sA[buf][m * BK + half * 8 + 16];
            #pragma unroll
            for (int e = 0; e < 8; ++e) { afrag[i][e] = lo[e]; afrag[i][e + 8] = hi[e]; }
        }
        // B 32x16 frag: lane half g holds K = g*8+{0..7} and 16+g*8+{0..7}, N = l15
        v16bf bfrag[4];
        #pragma unroll
        for (int j = 0; j < 4; ++j) {
            int n = wave_n + j * 16 + l15;
            v8bf lo = *(const v8bf*)&sB[buf][n * BK + half * 8];
            v8bf hi = *(const v8bf*)&sB[buf][n * BK + half * 8 + 16];
            #pragma unroll
            for (int e = 0; e < 8; ++e) { bfrag[j][e] = lo[e]; bfrag[j][e + 8] = hi[e]; }
        }
        #pragma unroll
        for (int i = 0; i < 2; ++i)
            #pragma unroll
            for (int j = 0; j < 4; ++j)
                acc[i][j] = __builtin_amdgcn_wmma_f32_16x16x32_bf16(
                    false, afrag[i], false, bfrag[j],
                    (short)0, acc[i][j], false, false);
    };

    stage(0, 0);                             // prologue: tile 0 in flight
    int kb = 0;
    int k0 = 0;
    // branchless steady state: stage next tile, wait only on current, compute
    for (; k0 < D - BK; k0 += BK, kb ^= 1) {
        stage(kb ^ 1, k0 + BK);
#if ASYNC_STAGE
        wait_asynccnt<4>();                  // current tile landed; next in flight
#endif
        __syncthreads();
        compute(kb);
        __syncthreads();                     // reads done before buffer is reused
    }
    // peeled last K-step
#if ASYNC_STAGE
    wait_asynccnt<0>();
#endif
    __syncthreads();
    compute(kb);

    // --- epilogue: local = sum(acc * diff) over this wave's 32x64 region ---
    // C layout: lanes 0-15 -> N=l15, M=v; lanes 16-31 -> N=l15, M=8+v
    float local = 0.0f;
    #pragma unroll
    for (int i = 0; i < 2; ++i) {
        #pragma unroll
        for (int j = 0; j < 4; ++j) {
            int n     = block_n + wave_n + j * 16 + l15;
            int mbase = block_m + wave_m + i * 16 + half * 8;
            #pragma unroll
            for (int v = 0; v < 8; ++v) {
                float d = (float)A[(size_t)(mbase + v) * D + n];
                local += acc[i][j][v] * d;
            }
        }
    }
    #pragma unroll
    for (int off = 16; off > 0; off >>= 1) local += __shfl_xor(local, off, 32);
    __shared__ float wsum[8];
    if (lane == 0) wsum[wid] = local;
    __syncthreads();
    if (tid == 0) {
        float s = 0.0f;
        #pragma unroll
        for (int w = 0; w < 8; ++w) s += wsum[w];
        partial[blockIdx.y * gridDim.x + blockIdx.x] = s;
    }
}

// ---------------------------------------------------------------------------
// Finalize: deterministically reduce all partial arrays, emit the 3 outputs.
// ---------------------------------------------------------------------------
__global__ __launch_bounds__(256) void rdl_finalize_kernel(const float* __restrict__ quadp, int nq,
                                                           const float* __restrict__ logp,  int nl,
                                                           const float* __restrict__ tracep, int nt,
                                                           float* __restrict__ out,
                                                           float scale, float bpp_factor) {
    int tid = threadIdx.x;
    float q = 0.0f, l = 0.0f, t = 0.0f;
    for (int i = tid; i < nq; i += 256) q += quadp[i];
    for (int i = tid; i < nl; i += 256) l += logp[i];
    for (int i = tid; i < nt; i += 256) t += tracep[i];
    #pragma unroll
    for (int off = 16; off > 0; off >>= 1) {
        q += __shfl_xor(q, off, 32);
        l += __shfl_xor(l, off, 32);
        t += __shfl_xor(t, off, 32);
    }
    __shared__ float sq[8], sl[8], st[8];
    int lane = tid & 31, wid = tid >> 5;
    if (lane == 0) { sq[wid] = q; sl[wid] = l; st[wid] = t; }
    __syncthreads();
    if (tid == 0) {
        float Q = 0.0f, L = 0.0f, T = 0.0f;
        #pragma unroll
        for (int w = 0; w < 8; ++w) { Q += sq[w]; L += sl[w]; T += st[w]; }
        float c = (T > 0.0f) ? (scale / T) : 1.0f;   // Hn = c*H folded into scalar
        float adaptive = Q * c;                      // / STD^2 with STD = 1
        float bpp = L * bpp_factor;                  // = sum(log) / (-ln2 * numel)
        out[0] = adaptive + bpp;                     // LMBDA = 1
        out[1] = adaptive;
        out[2] = bpp;
    }
}

// ---------------------------------------------------------------------------
// Host-side launch
// ---------------------------------------------------------------------------
extern "C" void kernel_launch(void* const* d_in, const int* in_sizes, int n_in,
                              void* d_out, int out_size, void* d_ws, size_t ws_size,
                              hipStream_t stream) {
    const float* ori_W = (const float*)d_in[0];
    const float* x_hat = (const float*)d_in[1];
    // d_in[2] = x : only numel is used
    const float* lik   = (const float*)d_in[3];
    const float* H     = (const float*)d_in[4];

    const size_t ND = (size_t)in_sizes[0];          // N*D
    const size_t DD = (size_t)in_sizes[4];          // D*D
    const int    D  = (int)lround(sqrt((double)DD));
    const int    N  = (int)(ND / (size_t)D);
    const size_t numel = (size_t)in_sizes[2];       // x.size

    char* ws = (char*)d_ws;
    __bf16* dA  = (__bf16*)ws;                      // diff bf16 [N,D]
    __bf16* dBt = (__bf16*)(ws + ND * 2);           // H^T bf16 [D,D]
    const int nQuad  = (N / BM) * (D / BN);
    const int nLog   = (int)(ND / 1024);
    const int nTrace = (D + 255) / 256;
    float* quadp  = (float*)(ws + ND * 2 + DD * 2);
    float* logp   = quadp + nQuad;
    float* tracep = logp + nLog;

    rdl_trace_kernel<<<nTrace, 256, 0, stream>>>(H, tracep, D);
    rdl_logsum_kernel<<<nLog, 256, 0, stream>>>(lik, logp);
    rdl_diff_bf16_kernel<<<(int)(ND / 1024), 256, 0, stream>>>(x_hat, ori_W, dA);
    rdl_transpose_bf16_kernel<<<dim3(D / 32, D / 32), dim3(32, 8), 0, stream>>>(H, dBt, D);

    rdl_gemm_quad_kernel<<<dim3(N / BM, D / BN), 256, 0, stream>>>(dA, dBt, quadp, D);

    const float scale = (float)D / (float)ND;                        // = 1/N
    const float bpp_factor = (float)(-1.0 / (log(2.0) * (double)numel));
    rdl_finalize_kernel<<<1, 256, 0, stream>>>(quadp, nQuad, logp, nLog,
                                               tracep, nTrace,
                                               (float*)d_out, scale, bpp_factor);
}
